// Attention_65738769432984
// MI455X (gfx1250) — matrix operand order repaired and checked by reading, not compiled, not run on hardware
//
#include <hip/hip_runtime.h>
#include <hip/hip_bf16.h>

typedef __attribute__((ext_vector_type(16))) __bf16 v16bf;
typedef __attribute__((ext_vector_type(8)))  __bf16 v8bf;
typedef __attribute__((ext_vector_type(8)))  float  v8f;

#define WMMA_BF16(A, B, C) \
  __builtin_amdgcn_wmma_f32_16x16x32_bf16(false, (A), false, (B), (short)0, (C), false, false)

static __device__ __forceinline__ v16bf cat8(v8bf lo, v8bf hi) {
  return __builtin_shufflevector(lo, hi, 0, 1, 2, 3, 4, 5, 6, 7, 8, 9, 10, 11, 12, 13, 14, 15);
}

// ---------------------------------------------------------------------------
// Generic row LayerNorm (+optional SiLU).
// mode 0: v = X[row,j];                 LN
// mode 1: v = pos(row)*wvec[j]+bias[j]; LN; SiLU   (rel-pos MLP layer 0)
// mode 2: v = X[row,j]+bias[j];         LN; SiLU   (rel-pos MLP layer 1)
// ---------------------------------------------------------------------------
__global__ __launch_bounds__(256)
void ln_act_kernel(const float* __restrict__ X, const float* __restrict__ wvec,
                   const float* __restrict__ bias, const float* __restrict__ gamma,
                   unsigned short* out_bf, float* out_f32, int cols, int mode) {
  __shared__ float s1[256], s2[256];
  const int row = blockIdx.x;
  const int tid = threadIdx.x;
  const float pos = (float)(row - 2047);
  float sum = 0.f, sumsq = 0.f;
  for (int j = tid; j < cols; j += 256) {
    float v;
    if (mode == 1)      v = pos * wvec[j] + bias[j];
    else if (mode == 2) v = X[(size_t)row * cols + j] + bias[j];
    else                v = X[(size_t)row * cols + j];
    sum += v; sumsq += v * v;
  }
  s1[tid] = sum; s2[tid] = sumsq;
  __syncthreads();
  for (int s = 128; s > 0; s >>= 1) {
    if (tid < s) { s1[tid] += s1[tid + s]; s2[tid] += s2[tid + s]; }
    __syncthreads();
  }
  const float mean = s1[0] / (float)cols;
  const float var  = s2[0] / (float)cols - mean * mean;
  const float rstd = rsqrtf(fmaxf(var, 0.f) + 1e-5f);
  __bf16* ob = reinterpret_cast<__bf16*>(out_bf);
  for (int j = tid; j < cols; j += 256) {
    float v;
    if (mode == 1)      v = pos * wvec[j] + bias[j];
    else if (mode == 2) v = X[(size_t)row * cols + j] + bias[j];
    else                v = X[(size_t)row * cols + j];
    float y = (v - mean) * rstd * gamma[j];
    if (mode != 0) y = y * (1.f / (1.f + __expf(-y)));   // SiLU
    if (out_bf) ob[(size_t)row * cols + j] = (__bf16)y;
    else        out_f32[(size_t)row * cols + j] = y;
  }
}

// ---------------------------------------------------------------------------
// f32 (KxN row-major) -> bf16 transposed (NxK row-major)
// ---------------------------------------------------------------------------
__global__ void cvt_t_bf16_kernel(const float* __restrict__ src, unsigned short* dst,
                                  int K, int N) {
  const int i = blockIdx.x * blockDim.x + threadIdx.x;
  if (i >= K * N) return;
  const int k = i / N, n = i % N;
  reinterpret_cast<__bf16*>(dst)[(size_t)n * K + k] = (__bf16)src[i];
}

// ---------------------------------------------------------------------------
// bf16 WMMA GEMM: C(MxN,f32) = A(MxK bf16 row-major) * Bt(NxK bf16 row-major)^T
// One wave32 per 16x16 tile; K stepped by 32. Software-pipelined: fragment
// k+32 is loaded before the WMMA on fragment k issues, so the WMMA only needs
// a partial loadcnt wait. All loads are b128, all addresses clamped in-bounds.
// ---------------------------------------------------------------------------
__global__ __launch_bounds__(32)
void gemm_bf16_wmma_kernel(const unsigned short* __restrict__ Au,
                           const unsigned short* __restrict__ Btu,
                           float* __restrict__ C, int M, int N, int K) {
  const __bf16* A  = reinterpret_cast<const __bf16*>(Au);
  const __bf16* Bt = reinterpret_cast<const __bf16*>(Btu);
  const int ntn = N >> 4;
  const int tm = blockIdx.x / ntn, tn = blockIdx.x % ntn;
  const int m0 = tm << 4, n0 = tn << 4;
  const int lane  = threadIdx.x & 31;
  const int abase = (lane >= 16) ? 16 : 0;
  const int bbase = (lane >= 16) ? 16 : 0;
  const int am    = min(m0 + (lane & 15), M - 1);       // clamp: OOB rows never stored
  const __bf16* arow = A  + (size_t)am * K;
  const __bf16* brow = Bt + (size_t)(n0 + (lane & 15)) * K;

  // prologue: load chunk 0
  v16bf a_cur = cat8(*reinterpret_cast<const v8bf*>(arow + abase),
                     *reinterpret_cast<const v8bf*>(arow + 8 + abase));
  v16bf b_cur = *reinterpret_cast<const v16bf*>(brow + bbase);
  v8f acc = {};
  for (int k0 = 0; k0 < K; k0 += 32) {
    const int kn = min(k0 + 32, K - 32);                // clamped next-chunk base
    const int kp = min(k0 + 64, K - 32);                // clamped prefetch base
    __builtin_prefetch(arow + kp, 0, 0);
    __builtin_prefetch(brow + kp, 0, 0);
    v16bf a_nxt = cat8(*reinterpret_cast<const v8bf*>(arow + kn + abase),
                       *reinterpret_cast<const v8bf*>(arow + kn + 8 + abase));
    v16bf b_nxt = *reinterpret_cast<const v16bf*>(brow + kn + bbase);
    acc = WMMA_BF16(a_cur, b_cur, acc);
    a_cur = a_nxt;
    b_cur = b_nxt;
  }
  const int hi = (lane >= 16) ? 8 : 0;
#pragma unroll
  for (int r = 0; r < 8; ++r) {
    const int row = m0 + r + hi;
    if (row < M) C[(size_t)row * N + n0 + (lane & 15)] = acc[r];
  }
}

// ---------------------------------------------------------------------------
// q-hat: per (b,i,h): l2norm over 64 dims * q_scale * SCALE(=8); (b,h,n,64) bf16
// ---------------------------------------------------------------------------
__global__ __launch_bounds__(32)
void qnorm_kernel(const float* __restrict__ qraw, const float* __restrict__ q_scale,
                  unsigned short* qhat_u) {
  const int gid = blockIdx.x;                 // 2*2048*16 blocks
  const int h = gid & 15, i = (gid >> 4) & 2047, b = gid >> 15;
  const int lane = threadIdx.x;
  const float* src = qraw + ((size_t)b * 2048 + i) * 1024 + h * 64;
  float x0 = src[lane], x1 = src[lane + 32];
  float ss = x0 * x0 + x1 * x1;
#pragma unroll
  for (int off = 16; off >= 1; off >>= 1) ss += __shfl_xor(ss, off, 32);
  const float inv = 8.0f / fmaxf(sqrtf(ss), 1e-12f);   // SCALE folded in
  __bf16* qhat = reinterpret_cast<__bf16*>(qhat_u);
  const size_t base = (((size_t)b * 16 + h) * 2048 + i) * 64;
  qhat[base + lane]      = (__bf16)(x0 * inv * q_scale[lane]);
  qhat[base + lane + 32] = (__bf16)(x1 * inv * q_scale[lane + 32]);
}

// ---------------------------------------------------------------------------
// k/v: split kv, l2norm k * k_scale, prepend null row.
// kpad:  (b, 2049, 64) bf16 row-major       (contiguous per key -> wide loads)
// vpadT: (b, 64, 2080) bf16, V transposed   (contiguous per dim -> wide loads)
// Pad columns 2049..2079 of vpadT zeroed (feed P==0 columns; must be finite).
// ---------------------------------------------------------------------------
#define VSTRIDE 2080
__global__ __launch_bounds__(32)
void kvnorm_kernel(const float* __restrict__ kvraw, const float* __restrict__ k_scale,
                   const float* __restrict__ null_kv,
                   unsigned short* kpad_u, unsigned short* vpadT_u) {
  const int gid = blockIdx.x;                 // 2*2080 blocks
  const int b = gid / VSTRIDE, ip = gid % VSTRIDE;
  const int lane = threadIdx.x;
  __bf16* vpT = reinterpret_cast<__bf16*>(vpadT_u) + (size_t)b * 64 * VSTRIDE;
  if (ip >= 2049) {                           // zero the alignment padding
    vpT[(size_t)lane * VSTRIDE + ip]        = (__bf16)0.f;
    vpT[(size_t)(lane + 32) * VSTRIDE + ip] = (__bf16)0.f;
    return;
  }
  float k0, k1, v0, v1;
  if (ip == 0) {
    k0 = null_kv[lane];      k1 = null_kv[lane + 32];
    v0 = null_kv[64 + lane]; v1 = null_kv[64 + lane + 32];
  } else {
    const float* r = kvraw + ((size_t)b * 2048 + (ip - 1)) * 128;
    k0 = r[lane];      k1 = r[lane + 32];
    v0 = r[64 + lane]; v1 = r[64 + lane + 32];
  }
  float ss = k0 * k0 + k1 * k1;
#pragma unroll
  for (int off = 16; off >= 1; off >>= 1) ss += __shfl_xor(ss, off, 32);
  const float inv = 1.f / fmaxf(sqrtf(ss), 1e-12f);
  __bf16* kp = reinterpret_cast<__bf16*>(kpad_u);
  const size_t kb = ((size_t)b * 2049 + ip) * 64;
  kp[kb + lane]      = (__bf16)(k0 * inv * k_scale[lane]);
  kp[kb + lane + 32] = (__bf16)(k1 * inv * k_scale[lane + 32]);
  vpT[(size_t)lane * VSTRIDE + ip]        = (__bf16)v0;
  vpT[(size_t)(lane + 32) * VSTRIDE + ip] = (__bf16)v1;
}

// ---------------------------------------------------------------------------
// rel[i][h] = h1[i,:] @ w2[:,h] + b2[h]   (4095 x 16, K=1024)
// ---------------------------------------------------------------------------
__global__ void rel_out_kernel(const unsigned short* __restrict__ h1u,
                               const float* __restrict__ w2,
                               const float* __restrict__ b2, float* __restrict__ rel) {
  const int idx = blockIdx.x * blockDim.x + threadIdx.x;
  if (idx >= 4095 * 16) return;
  const int i = idx >> 4, hh = idx & 15;
  const __bf16* h1 = reinterpret_cast<const __bf16*>(h1u);
  float s = b2[hh];
  for (int k = 0; k < 1024; ++k) s += (float)h1[(size_t)i * 1024 + k] * w2[k * 16 + hh];
  rel[idx] = s;
}

// ---------------------------------------------------------------------------
// Flash attention: one wave32 per (b, h, 16-query tile).
// 32 padded keys / iter: QK^T (4 WMMA) -> online softmax -> P via LDS ->
// PV (4 WMMA). All operand loads are contiguous b128s via clamped addressing;
// masking (causal + pad) is applied arithmetically afterwards.
// ---------------------------------------------------------------------------
__global__ __launch_bounds__(32)
void attn_kernel(const unsigned short* __restrict__ qhat_u,
                 const unsigned short* __restrict__ kpad_u,
                 const unsigned short* __restrict__ vpadT_u,
                 const float* __restrict__ rel, const float* __restrict__ null_bias,
                 unsigned short* __restrict__ out_u) {
  __shared__ __bf16 pt[16 * 32];
  const __bf16* qhat = reinterpret_cast<const __bf16*>(qhat_u);
  const __bf16* kpad = reinterpret_cast<const __bf16*>(kpad_u);
  const __bf16* vpT  = reinterpret_cast<const __bf16*>(vpadT_u);
  __bf16* out = reinterpret_cast<__bf16*>(out_u);

  const int blk = blockIdx.x;                  // 2*16*128 blocks
  const int qt = blk & 127, h = (blk >> 7) & 15, b = blk >> 11;
  const int qi0 = qt << 4;
  const int lane  = threadIdx.x;
  const int coln  = lane & 15;
  const int abase = (lane >= 16) ? 16 : 0;
  const int bbase = (lane >= 16) ? 16 : 0;
  const int hi    = (lane >= 16) ? 8 : 0;

  const __bf16* Kp  = kpad + (size_t)b * 2049 * 64;
  const __bf16* VpT = vpT + (size_t)b * 64 * VSTRIDE;

  // Q tile as two A fragments (d 0..31 and 32..63): four b128 loads
  const __bf16* qr = qhat + (((size_t)b * 16 + h) * 2048 + qi0 + coln) * 64;
  v16bf qa0 = cat8(*reinterpret_cast<const v8bf*>(qr + abase),
                   *reinterpret_cast<const v8bf*>(qr + 8 + abase));
  v16bf qa1 = cat8(*reinterpret_cast<const v8bf*>(qr + 32 + abase),
                   *reinterpret_cast<const v8bf*>(qr + 40 + abase));

  float mrow[8], lrow[8];
  v8f o[4] = {};
#pragma unroll
  for (int r = 0; r < 8; ++r) { mrow[r] = -3.0e38f; lrow[r] = 0.f; }

  const float nb = null_bias[h];
  const int nch = (qi0 + 16) / 32 + 1;          // covers padded j <= qi0+16

  for (int c = 0; c < nch; ++c) {
    const int j0 = c * 32;
    v8f sv[2];
#pragma unroll
    for (int t = 0; t < 2; ++t) {
      const int jn = j0 + t * 16 + coln;        // key column this lane supplies
      const __bf16* kr = Kp + (size_t)min(jn, 2048) * 64;   // clamp; masked below
      v16bf kb0 = *reinterpret_cast<const v16bf*>(kr + bbase);
      v16bf kb1 = *reinterpret_cast<const v16bf*>(kr + 32 + bbase);
      v8f st = {};
      st = WMMA_BF16(qa0, kb0, st);
      st = WMMA_BF16(qa1, kb1, st);
#pragma unroll
      for (int r = 0; r < 8; ++r) {
        const int i = qi0 + r + hi;
        const int j = j0 + t * 16 + coln;
        int idx = i - j + 2048;
        idx = min(max(idx, 0), 4094);           // clamp; wrong values are masked
        const float rb = rel[(size_t)idx * 16 + h];
        const float bias = (j == 0) ? nb : rb;
        const bool ok = (j < 2049) && (j <= i + 1);
        sv[t][r] = ok ? (st[r] + bias) : -3.0e38f;
      }
    }
    // online softmax over these 32 columns; stash P (bf16) in LDS
#pragma unroll
    for (int r = 0; r < 8; ++r) {
      float mx = fmaxf(sv[0][r], sv[1][r]);
#pragma unroll
      for (int off = 8; off >= 1; off >>= 1) mx = fmaxf(mx, __shfl_xor(mx, off, 16));
      const float mnew  = fmaxf(mrow[r], mx);
      const float alpha = __expf(mrow[r] - mnew);
      mrow[r] = mnew;
      const float p0 = __expf(sv[0][r] - mnew);
      const float p1 = __expf(sv[1][r] - mnew);
      float ps = p0 + p1;
#pragma unroll
      for (int off = 8; off >= 1; off >>= 1) ps += __shfl_xor(ps, off, 16);
      lrow[r] = lrow[r] * alpha + ps;
      o[0][r] *= alpha; o[1][r] *= alpha; o[2][r] *= alpha; o[3][r] *= alpha;
      pt[(r + hi) * 32 + coln]      = (__bf16)p0;
      pt[(r + hi) * 32 + 16 + coln] = (__bf16)p1;
    }
    __syncthreads();
    // re-read P as an A fragment (two ds_load_b128)
    v16bf pa = cat8(*reinterpret_cast<const v8bf*>(pt + coln * 32 + abase),
                    *reinterpret_cast<const v8bf*>(pt + coln * 32 + 8 + abase));
    __syncthreads();
    // P (16x32) @ V (32x64) as 4 N-tiles; V^T rows are contiguous in j
#pragma unroll
    for (int dt = 0; dt < 4; ++dt) {
      const __bf16* vrow = VpT + (size_t)(dt * 16 + coln) * VSTRIDE + j0 + bbase;
      v16bf vb = *reinterpret_cast<const v16bf*>(vrow);
      o[dt] = WMMA_BF16(pa, vb, o[dt]);
    }
  }
  // epilogue: divide by row sums, store bf16 in (b, n, h*64+d)
#pragma unroll
  for (int r = 0; r < 8; ++r) {
    const float inv = 1.f / lrow[r];
    const int row = qi0 + r + hi;
    const size_t base = ((size_t)b * 2048 + row) * 1024 + h * 64;
#pragma unroll
    for (int dt = 0; dt < 4; ++dt)
      out[base + dt * 16 + coln] = (__bf16)(o[dt][r] * inv);
  }
}

// ---------------------------------------------------------------------------
// Host-side orchestration
// ---------------------------------------------------------------------------
extern "C" void kernel_launch(void* const* d_in, const int* in_sizes, int n_in,
                              void* d_out, int out_size, void* d_ws, size_t ws_size,
                              hipStream_t stream) {
  (void)in_sizes; (void)n_in; (void)out_size; (void)ws_size;
  const float* x         = (const float*)d_in[0];
  /* d_in[1] = mask: all-true in this problem; padded col is always true */
  const float* g_norm    = (const float*)d_in[2];
  const float* Wq        = (const float*)d_in[3];
  const float* Wkv       = (const float*)d_in[4];
  const float* q_scale   = (const float*)d_in[5];
  const float* k_scale   = (const float*)d_in[6];
  const float* null_kv   = (const float*)d_in[7];
  const float* null_bias = (const float*)d_in[8];
  const float* w0        = (const float*)d_in[9];
  const float* b0        = (const float*)d_in[10];
  const float* g0        = (const float*)d_in[11];
  const float* w1        = (const float*)d_in[12];
  const float* b1        = (const float*)d_in[13];
  const float* g1        = (const float*)d_in[14];
  const float* w2        = (const float*)d_in[15];
  const float* b2        = (const float*)d_in[16];
  const float* Wout      = (const float*)d_in[17];
  const float* g_out     = (const float*)d_in[18];
  float* out = (float*)d_out;

  char* ws = (char*)d_ws;
  size_t off = 0;
  auto alloc = [&](size_t bytes) -> char* {
    char* p = ws + off; off += (bytes + 255) & ~(size_t)255; return p;
  };
  // region A: xn (bf16), later reused as h1 (bf16)
  unsigned short* xn_bf   = (unsigned short*)alloc(4096ull * 1024 * 2);
  unsigned short* wq_t    = (unsigned short*)alloc(1024ull * 1024 * 2);   // (N=1024) x (K=1024)
  unsigned short* wkv_t   = (unsigned short*)alloc(128ull * 1024 * 2);    // (N=128)  x (K=1024)
  unsigned short* w1_t    = (unsigned short*)alloc(1024ull * 1024 * 2);
  unsigned short* wout_t  = (unsigned short*)alloc(1024ull * 1024 * 2);
  // region F: qraw -> h1raw -> yraw (f32)
  float*          fbufF   = (float*)alloc(4096ull * 1024 * 4);
  float*          kvraw   = (float*)alloc(4096ull * 128 * 4);
  unsigned short* qhat    = (unsigned short*)alloc(2ull * 16 * 2048 * 64 * 2);
  unsigned short* kpad    = (unsigned short*)alloc(2ull * 2049 * 64 * 2);
  unsigned short* vpadT   = (unsigned short*)alloc(2ull * 64 * VSTRIDE * 2);
  unsigned short* h0_bf   = (unsigned short*)alloc(4096ull * 1024 * 2);
  float*          rel     = (float*)alloc(4095ull * 16 * 4);
  unsigned short* ao_bf   = (unsigned short*)alloc(4096ull * 1024 * 2);
  unsigned short* h1_bf   = xn_bf;   // reuse: xn dead after projections

  // 1) xn = LN(x) * g_norm  -> bf16
  ln_act_kernel<<<4096, 256, 0, stream>>>(x, nullptr, nullptr, g_norm, xn_bf, nullptr, 1024, 0);
  // 2) weights -> bf16, transposed to NxK for contiguous B fragments
  cvt_t_bf16_kernel<<<4096, 256, 0, stream>>>(Wq, wq_t, 1024, 1024);
  cvt_t_bf16_kernel<<<512,  256, 0, stream>>>(Wkv, wkv_t, 1024, 128);
  cvt_t_bf16_kernel<<<4096, 256, 0, stream>>>(w1, w1_t, 1024, 1024);
  cvt_t_bf16_kernel<<<4096, 256, 0, stream>>>(Wout, wout_t, 1024, 1024);
  // 3) q = xn @ Wq ; kv = xn @ Wkv
  gemm_bf16_wmma_kernel<<<256 * 64, 32, 0, stream>>>(xn_bf, wq_t, fbufF, 4096, 1024, 1024);
  gemm_bf16_wmma_kernel<<<256 * 8,  32, 0, stream>>>(xn_bf, wkv_t, kvraw, 4096, 128, 1024);
  // 4) normalize q / k, prepend null kv, build V^T
  qnorm_kernel<<<2 * 2048 * 16, 32, 0, stream>>>(fbufF, q_scale, qhat);
  kvnorm_kernel<<<2 * VSTRIDE, 32, 0, stream>>>(kvraw, k_scale, null_kv, kpad, vpadT);
  // 5) rel-pos MLP: h0 = SiLU(LN(pos*w0+b0)) ; h1 = SiLU(LN(h0@w1+b1)) ; rel = h1@w2+b2
  ln_act_kernel<<<4095, 256, 0, stream>>>(w0, w0, b0, g0, h0_bf, nullptr, 1024, 1);
  gemm_bf16_wmma_kernel<<<256 * 64, 32, 0, stream>>>(h0_bf, w1_t, fbufF, 4095, 1024, 1024);
  ln_act_kernel<<<4095, 256, 0, stream>>>(fbufF, nullptr, b1, g1, h1_bf, nullptr, 1024, 2);
  rel_out_kernel<<<(4095 * 16 + 255) / 256, 256, 0, stream>>>(h1_bf, w2, b2, rel);
  // 6) flash attention
  attn_kernel<<<2 * 16 * 128, 32, 0, stream>>>(qhat, kpad, vpadT, rel, null_bias, ao_bf);
  // 7) y = attn_out @ Wout ; out = LN(y) * g_out
  gemm_bf16_wmma_kernel<<<256 * 64, 32, 0, stream>>>(ao_bf, wout_t, fbufF, 4096, 1024, 1024);
  ln_act_kernel<<<4096, 256, 0, stream>>>(fbufF, nullptr, nullptr, g_out, nullptr, out, 1024, 0);
}
